// TeacherSSM_28114855920377
// MI455X (gfx1250) — matrix operand order repaired and checked by reading, not compiled, not run on hardware
//
#include <hip/hip_runtime.h>
#include <hip/hip_bf16.h>
#include <math.h>

typedef __attribute__((ext_vector_type(16))) _Float16 v16h;
typedef __attribute__((ext_vector_type(8)))  _Float16 v8h;
typedef __attribute__((ext_vector_type(8)))  float    v8f;

#define D_MODEL   128
#define N_LAYERS  8
#define D_STATE   16
#define D_CONV    4
#define STRIDE    4
#define N_CLASSES 35
#define D_INNER   256
#define DT_RANK   8
#define BATCH     4
#define SEQ       2048
#define ROWS      (BATCH * SEQ)            // 8192
#define DBL_W     (DT_RANK + 2 * D_STATE)  // 40 (logical)
#define DBL_P     48                       // padded dbl row stride
#define DTW_KP    32                       // padded dt_w K

// ---------------------------------------------------------------------------
// Stem: non-overlapping frames (B, L, 4) x stem_w (128,4) -> h (B,L,128)
// ---------------------------------------------------------------------------
__global__ void stem_kernel(const float* __restrict__ x,
                            const float* __restrict__ w,
                            const float* __restrict__ bias,
                            float* __restrict__ h) {
  int idx = blockIdx.x * blockDim.x + threadIdx.x;
  if (idx >= ROWS * D_MODEL) return;
  int d = idx % D_MODEL;
  int row = idx / D_MODEL;               // b*SEQ + l ; frame starts at row*4
  const float4 fr = *(const float4*)(x + (size_t)row * STRIDE);
  const float4 wd = *(const float4*)(w + d * STRIDE);
  h[idx] = bias[d] + fr.x * wd.x + fr.y * wd.y + fr.z * wd.z + fr.w * wd.w;
}

// ---------------------------------------------------------------------------
// One-time weight conversion / padding (f32 -> f16, exact tiling).
// ---------------------------------------------------------------------------
__global__ void cvt_f16_kernel(const float* __restrict__ in,
                               _Float16* __restrict__ outp, int n) {
  int i = blockIdx.x * blockDim.x + threadIdx.x;
  if (i < n) outp[i] = (_Float16)in[i];
}

// x_proj_w (8,40,256) -> f16 (8,48,256), zero rows 40..47
__global__ void pad_xproj_kernel(const float* __restrict__ in, _Float16* __restrict__ outp) {
  int idx = blockIdx.x * blockDim.x + threadIdx.x;      // 8*48*256
  if (idx >= N_LAYERS * DBL_P * D_INNER) return;
  int k  = idx % D_INNER;
  int r  = (idx / D_INNER) % DBL_P;
  int li = idx / (D_INNER * DBL_P);
  outp[idx] = (r < DBL_W) ? (_Float16)in[((size_t)li * DBL_W + r) * D_INNER + k]
                          : (_Float16)0.f;
}

// dt_w (8,256,8) -> f16 (8,256,32), zero cols 8..31
__global__ void pad_dtw_kernel(const float* __restrict__ in, _Float16* __restrict__ outp) {
  int idx = blockIdx.x * blockDim.x + threadIdx.x;      // 8*256*32
  if (idx >= N_LAYERS * D_INNER * DTW_KP) return;
  int k  = idx % DTW_KP;
  int n  = (idx / DTW_KP) % D_INNER;
  int li = idx / (DTW_KP * D_INNER);
  outp[idx] = (k < DT_RANK) ? (_Float16)in[((size_t)li * D_INNER + n) * DT_RANK + k]
                            : (_Float16)0.f;
}

// ---------------------------------------------------------------------------
// LayerNorm over last dim (128); emits f16 (consumed only by in_proj GEMM).
// ---------------------------------------------------------------------------
__global__ void ln_kernel(const float* __restrict__ h,
                          const float* __restrict__ g,
                          const float* __restrict__ b,
                          _Float16* __restrict__ xn) {
  __shared__ float red[D_MODEL];
  int row = blockIdx.x;
  int t = threadIdx.x;
  float v = h[(size_t)row * D_MODEL + t];
  red[t] = v;
  __syncthreads();
  for (int s = 64; s > 0; s >>= 1) { if (t < s) red[t] += red[t + s]; __syncthreads(); }
  float mu = red[0] * (1.0f / D_MODEL);
  __syncthreads();
  float dv = v - mu;
  red[t] = dv * dv;
  __syncthreads();
  for (int s = 64; s > 0; s >>= 1) { if (t < s) red[t] += red[t + s]; __syncthreads(); }
  float var = red[0] * (1.0f / D_MODEL);
  xn[(size_t)row * D_MODEL + t] = (_Float16)(dv * rsqrtf(var + 1e-5f) * g[t] + b[t]);
}

// ---------------------------------------------------------------------------
// Fragment loaders (wave32; contiguous K order k=16h+s, same in A and B).
// A (16x32, MxK): lane<16 row=lane holds K kb..kb+15 (kb=0),
//                 lane>=16 same rows with kb=16.
// B (32x16, KxN): lane%16 = column, halves are 16 contiguous K values.
// ---------------------------------------------------------------------------
__device__ inline v16h load_a_frag(const float* Arow, int k0, int kbA) {
  const float4 a0 = *(const float4*)(Arow + k0 + kbA);
  const float4 a1 = *(const float4*)(Arow + k0 + kbA + 4);
  const float4 a2 = *(const float4*)(Arow + k0 + kbA + 8);
  const float4 a3 = *(const float4*)(Arow + k0 + kbA + 12);
  v16h a;
  a[0]  = (_Float16)a0.x; a[1]  = (_Float16)a0.y; a[2]  = (_Float16)a0.z; a[3]  = (_Float16)a0.w;
  a[4]  = (_Float16)a1.x; a[5]  = (_Float16)a1.y; a[6]  = (_Float16)a1.z; a[7]  = (_Float16)a1.w;
  a[8]  = (_Float16)a2.x; a[9]  = (_Float16)a2.y; a[10] = (_Float16)a2.z; a[11] = (_Float16)a2.w;
  a[12] = (_Float16)a3.x; a[13] = (_Float16)a3.y; a[14] = (_Float16)a3.z; a[15] = (_Float16)a3.w;
  return a;
}

__device__ inline v16h load_a_frag(const _Float16* Arow, int k0, int kbA) {
  const v8h lo = *(const v8h*)(Arow + k0 + kbA);        // K kb..kb+7
  const v8h hi = *(const v8h*)(Arow + k0 + kbA + 8);    // K kb+8..kb+15
  v16h a;
#pragma unroll
  for (int i = 0; i < 8; ++i) { a[i] = lo[i]; a[i + 8] = hi[i]; }
  return a;
}

__device__ inline v16h load_b_frag(const float* Wrow, int k0, int kbB) {
  const float4 b0 = *(const float4*)(Wrow + k0 + kbB);
  const float4 b1 = *(const float4*)(Wrow + k0 + kbB + 4);
  const float4 b2 = *(const float4*)(Wrow + k0 + kbB + 8);
  const float4 b3 = *(const float4*)(Wrow + k0 + kbB + 12);
  v16h b;
  b[0]  = (_Float16)b0.x; b[1]  = (_Float16)b0.y; b[2]  = (_Float16)b0.z; b[3]  = (_Float16)b0.w;
  b[4]  = (_Float16)b1.x; b[5]  = (_Float16)b1.y; b[6]  = (_Float16)b1.z; b[7]  = (_Float16)b1.w;
  b[8]  = (_Float16)b2.x; b[9]  = (_Float16)b2.y; b[10] = (_Float16)b2.z; b[11] = (_Float16)b2.w;
  b[12] = (_Float16)b3.x; b[13] = (_Float16)b3.y; b[14] = (_Float16)b3.z; b[15] = (_Float16)b3.w;
  return b;
}

__device__ inline v16h load_b_frag(const _Float16* Wrow, int k0, int kbB) {
  return *(const v16h*)(Wrow + k0 + kbB);               // 16 contiguous K halves
}

// ---------------------------------------------------------------------------
// WMMA GEMM: C[M,N] = A[M,K(LDA)] * W[N,K]^T   (f16 WMMA, f32 accumulate)
// Fully regular tiling (K%32==0, N%16==0, exact grid): straight-line body.
// Per K-step: load A fragment + ALL NT B fragments first (one big VMEM
// clause), then issue the NT WMMAs — each WMMA waits only for its own B
// fragment (decreasing loadcnt), overlapping compute with in-flight loads.
// MODE 0: store   MODE 1: C += (residual)   MODE 2: softplus(v + bias[n])
// ---------------------------------------------------------------------------
template <typename TA, typename TB, int K, int N, int LDA, int LDC,
          int MODE, int NT, int WPB>
__global__ void wmma_gemm_kernel(const TA* __restrict__ A,
                                 const TB* __restrict__ W,
                                 float* __restrict__ C,
                                 const float* __restrict__ bias) {
  static_assert(K % 32 == 0 && N % 16 == 0, "regular tiling only");
  constexpr int KSTEPS = K / 32;
  const int lane   = threadIdx.x & 31;
  const int wave   = threadIdx.x >> 5;
  const int ntbase = (blockIdx.x * WPB + wave) * NT;  // first N-tile of this wave
  const int m0     = blockIdx.y * 16;
  const int nidx   = lane & 15;
  const int kbA    = (lane < 16) ? 0 : 16;
  const int kbB    = (lane < 16) ? 0 : 16;
  const TA* Arow = A + (size_t)(m0 + nidx) * LDA;

  v8f acc[NT];
#pragma unroll
  for (int t = 0; t < NT; ++t) acc[t] = (v8f){};

#pragma unroll
  for (int ks = 0; ks < KSTEPS; ++ks) {
    const int k0 = ks * 32;
    const v16h a = load_a_frag(Arow, k0, kbA);
    v16h bfrag[NT];
#pragma unroll
    for (int t = 0; t < NT; ++t) {
      const TB* Wrow = W + (size_t)((ntbase + t) * 16 + nidx) * K;
      bfrag[t] = load_b_frag(Wrow, k0, kbB);
    }
#pragma unroll
    for (int t = 0; t < NT; ++t) {
      acc[t] = __builtin_amdgcn_wmma_f32_16x16x32_f16(false, a, false, bfrag[t],
                                                      (short)0, acc[t], false, false);
    }
  }

  const int mbase = m0 + ((lane < 16) ? 0 : 8);       // D: VGPR r -> M=r / M=r+8
#pragma unroll
  for (int t = 0; t < NT; ++t) {
    const int ncol = (ntbase + t) * 16 + nidx;
#pragma unroll
    for (int r = 0; r < 8; ++r) {
      const size_t idx = (size_t)(mbase + r) * LDC + ncol;
      const float v = acc[t][r];
      if (MODE == 1) {
        C[idx] += v;
      } else if (MODE == 2) {
        const float s = v + bias[ncol];
        C[idx] = (s > 20.f) ? s : log1pf(__expf(s));  // softplus
      } else {
        C[idx] = v;
      }
    }
  }
}

// ---------------------------------------------------------------------------
// Causal depthwise conv (4 taps) + SiLU.  Emits f32 (scan) and f16 (GEMM A).
// ---------------------------------------------------------------------------
__global__ void conv_silu_kernel(const float* __restrict__ xz,
                                 const float* __restrict__ cw,
                                 const float* __restrict__ cb,
                                 float* __restrict__ xp,
                                 _Float16* __restrict__ xph) {
  int idx = blockIdx.x * blockDim.x + threadIdx.x;
  if (idx >= ROWS * D_INNER) return;
  int c = idx % D_INNER;
  int row = idx / D_INNER;
  int l = row % SEQ;
  int base = row - l;                    // batch base row
  const float4 wv = *(const float4*)(cw + c * D_CONV);
  float acc = cb[c];
  const float w4[4] = {wv.x, wv.y, wv.z, wv.w};
#pragma unroll
  for (int t = 0; t < D_CONV; ++t) {
    int ls = l - (D_CONV - 1) + t;
    if (ls >= 0)
      acc += w4[t] * xz[(size_t)(base + ls) * (2 * D_INNER) + c];
  }
  float s = acc / (1.f + __expf(-acc));  // silu
  xp[idx]  = s;
  xph[idx] = (_Float16)s;
}

// ---------------------------------------------------------------------------
// Selective scan. One block per batch, one thread per inner channel.
// 16 states in registers; Bm/Cm broadcast via LDS. Fuses +u*D and *silu(z).
// dbl rows are DBL_P(48)-wide: [0:8)=dt, [8:24)=Bm, [24:40)=Cm.
// Emits y as f16 (consumed only by out_proj GEMM).
// ---------------------------------------------------------------------------
__global__ void scan_kernel(const float* __restrict__ xp,
                            const float* __restrict__ delta,
                            const float* __restrict__ dbl,
                            const float* __restrict__ xz,
                            const float* __restrict__ A_log,
                            const float* __restrict__ Dv,
                            _Float16* __restrict__ y) {
  int b = blockIdx.x;
  int c = threadIdx.x;          // 0..255
  __shared__ float BC[32];      // [0:16)=Bm, [16:32)=Cm for current timestep

  float Arow[D_STATE];
#pragma unroll
  for (int n = 0; n < D_STATE; ++n) Arow[n] = -__expf(A_log[c * D_STATE + n]);
  float Dc = Dv[c];

  float st[D_STATE];
#pragma unroll
  for (int n = 0; n < D_STATE; ++n) st[n] = 0.f;

  for (int l = 0; l < SEQ; ++l) {
    size_t row = (size_t)b * SEQ + l;
    if (c < 32) BC[c] = dbl[row * DBL_P + DT_RANK + c];
    __syncthreads();

    float d  = delta[row * D_INNER + c];
    float u  = xp[row * D_INNER + c];
    float du = d * u;
    float yv = 0.f;
#pragma unroll
    for (int n = 0; n < D_STATE; ++n) {
      st[n] = __expf(d * Arow[n]) * st[n] + du * BC[n];
      yv += st[n] * BC[16 + n];
    }
    float z  = xz[row * (2 * D_INNER) + D_INNER + c];
    float sz = z / (1.f + __expf(-z));
    y[row * D_INNER + c] = (_Float16)((yv + u * Dc) * sz);
    __syncthreads();
  }
}

// ---------------------------------------------------------------------------
// Mean pool over L, then classifier head.
// ---------------------------------------------------------------------------
__global__ void pool_kernel(const float* __restrict__ h, float* __restrict__ pooled) {
  int b = blockIdx.x, d = threadIdx.x;   // blockDim = 128
  float s = 0.f;
  for (int l = 0; l < SEQ; ++l) s += h[((size_t)b * SEQ + l) * D_MODEL + d];
  pooled[b * D_MODEL + d] = s * (1.0f / SEQ);
}

__global__ void head_kernel(const float* __restrict__ pooled,
                            const float* __restrict__ hw,
                            const float* __restrict__ hb,
                            float* __restrict__ out) {
  int b = blockIdx.x, n = threadIdx.x;
  if (n >= N_CLASSES) return;
  float s = hb[n];
#pragma unroll 4
  for (int d = 0; d < D_MODEL; ++d) s += pooled[b * D_MODEL + d] * hw[n * D_MODEL + d];
  out[b * N_CLASSES + n] = s;
}

// ---------------------------------------------------------------------------
extern "C" void kernel_launch(void* const* d_in, const int* in_sizes, int n_in,
                              void* d_out, int out_size, void* d_ws, size_t ws_size,
                              hipStream_t stream) {
  (void)in_sizes; (void)n_in; (void)out_size; (void)ws_size;
  const float* x       = (const float*)d_in[0];
  const float* stem_w  = (const float*)d_in[1];
  const float* stem_b  = (const float*)d_in[2];
  const float* ln_g    = (const float*)d_in[3];
  const float* ln_b    = (const float*)d_in[4];
  const float* in_w    = (const float*)d_in[5];
  const float* conv_w  = (const float*)d_in[6];
  const float* conv_b  = (const float*)d_in[7];
  const float* xproj_w = (const float*)d_in[8];
  const float* dt_w    = (const float*)d_in[9];
  const float* dt_b    = (const float*)d_in[10];
  const float* A_log   = (const float*)d_in[11];
  const float* Dv      = (const float*)d_in[12];
  const float* out_w   = (const float*)d_in[13];
  const float* head_w  = (const float*)d_in[14];
  const float* head_b  = (const float*)d_in[15];
  float* out = (float*)d_out;

  // ---- workspace carve-up: f32 region first, then f16 region (~50 MB) ----
  float* w = (float*)d_ws;
  float* h_     = w;  w += (size_t)ROWS * D_MODEL;         // 1 M
  float* xz     = w;  w += (size_t)ROWS * 2 * D_INNER;     // 4 M
  float* xp     = w;  w += (size_t)ROWS * D_INNER;         // 2 M
  float* dblb   = w;  w += (size_t)ROWS * DBL_P;           // 0.39 M
  float* delta  = w;  w += (size_t)ROWS * D_INNER;         // 2 M
  float* pooled = w;  w += BATCH * D_MODEL;                // 512
  w += 8 - (BATCH * D_MODEL % 8);                          // keep 16B alignment

  _Float16* hw16 = (_Float16*)w;
  _Float16* xn_h    = hw16;  hw16 += (size_t)ROWS * D_MODEL;              // 1 M
  _Float16* xp_h    = hw16;  hw16 += (size_t)ROWS * D_INNER;              // 2 M
  _Float16* yb_h    = hw16;  hw16 += (size_t)ROWS * D_INNER;              // 2 M
  _Float16* in_w_h  = hw16;  hw16 += (size_t)N_LAYERS * 2 * D_INNER * D_MODEL;  // 0.5 M
  _Float16* out_w_h = hw16;  hw16 += (size_t)N_LAYERS * D_MODEL * D_INNER;      // 0.25 M
  _Float16* xw_h    = hw16;  hw16 += (size_t)N_LAYERS * DBL_P * D_INNER;        // 98 K
  _Float16* dtw_h   = hw16;  hw16 += (size_t)N_LAYERS * D_INNER * DTW_KP;       // 65 K

  // ---- one-time weight conversion/padding (all layers) ----
  {
    int n1 = N_LAYERS * 2 * D_INNER * D_MODEL;   // in_proj weights
    cvt_f16_kernel<<<(n1 + 255) / 256, 256, 0, stream>>>(in_w, in_w_h, n1);
    int n2 = N_LAYERS * D_MODEL * D_INNER;       // out_proj weights
    cvt_f16_kernel<<<(n2 + 255) / 256, 256, 0, stream>>>(out_w, out_w_h, n2);
    pad_xproj_kernel<<<(N_LAYERS * DBL_P * D_INNER + 255) / 256, 256, 0, stream>>>(xproj_w, xw_h);
    pad_dtw_kernel<<<(N_LAYERS * D_INNER * DTW_KP + 255) / 256, 256, 0, stream>>>(dt_w, dtw_h);
  }

  stem_kernel<<<(ROWS * D_MODEL + 255) / 256, 256, 0, stream>>>(x, stem_w, stem_b, h_);

  for (int li = 0; li < N_LAYERS; ++li) {
    ln_kernel<<<ROWS, D_MODEL, 0, stream>>>(h_, ln_g + li * D_MODEL, ln_b + li * D_MODEL, xn_h);

    // in_proj: (8192,128) x (512,128)^T -> xz (8192,512). 32 tiles = 2 x (4w x 4t).
    wmma_gemm_kernel<_Float16, _Float16, 128, 512, 128, 512, 0, 4, 4>
        <<<dim3(2, ROWS / 16), 128, 0, stream>>>(
            xn_h, in_w_h + (size_t)li * 512 * 128, xz, nullptr);

    conv_silu_kernel<<<(ROWS * D_INNER + 255) / 256, 256, 0, stream>>>(
        xz, conv_w + (size_t)li * D_INNER * D_CONV, conv_b + li * D_INNER, xp, xp_h);

    // x_proj: (8192,256) x (48,256)^T -> dbl (8192,48). 3 tiles = 1 x (3w x 1t).
    wmma_gemm_kernel<_Float16, _Float16, 256, DBL_P, 256, DBL_P, 0, 1, 3>
        <<<dim3(1, ROWS / 16), 96, 0, stream>>>(
            xp_h, xw_h + (size_t)li * DBL_P * D_INNER, dblb, nullptr);

    // delta = softplus(dbl[:,0:32] @ dtw^T + dt_b): K padded to 32, exact
    // (dbl cols 8..31 hold B/C values but multiply zero weight cols).
    wmma_gemm_kernel<float, _Float16, DTW_KP, 256, DBL_P, 256, 2, 4, 4>
        <<<dim3(1, ROWS / 16), 128, 0, stream>>>(
            dblb, dtw_h + (size_t)li * D_INNER * DTW_KP, delta, dt_b + li * D_INNER);

    scan_kernel<<<BATCH, D_INNER, 0, stream>>>(
        xp, delta, dblb, xz,
        A_log + (size_t)li * D_INNER * D_STATE, Dv + li * D_INNER, yb_h);

    // out_proj with fused residual: h += y @ out_w^T. 8 tiles = 1 x (4w x 2t).
    wmma_gemm_kernel<_Float16, _Float16, 256, 128, 256, 128, 1, 2, 4>
        <<<dim3(1, ROWS / 16), 128, 0, stream>>>(
            yb_h, out_w_h + (size_t)li * D_MODEL * D_INNER, h_, nullptr);
  }

  pool_kernel<<<BATCH, D_MODEL, 0, stream>>>(h_, pooled);
  head_kernel<<<BATCH, 64, 0, stream>>>(pooled, head_w, head_b, out);
}
